// GNN_86474871537797
// MI455X (gfx1250) — compile-verified, matrix operand order checked by reading, not run on hardware
//
#include <hip/hip_runtime.h>
#include <hip/hip_bf16.h>

typedef _Float16 half8 __attribute__((ext_vector_type(8)));
typedef _Float16 v16h  __attribute__((ext_vector_type(16)));
typedef float    v8f   __attribute__((ext_vector_type(8)));

#define D_IN   128
#define HC     256   // HEADS*HID
#define HID    64
#define HEADS  4
#define OUT_F  32
#define ET_N   32
#define SLOPE  0.2f
#define BN_EPS 1e-5f

// ---------- helpers ----------
static __device__ __forceinline__ unsigned f2ord(float f) {
    unsigned u = __float_as_uint(f);
    return (u & 0x80000000u) ? ~u : (u | 0x80000000u);
}
static __device__ __forceinline__ float ord2f(unsigned u) {
    unsigned v = (u & 0x80000000u) ? (u & 0x7FFFFFFFu) : ~u;
    return __uint_as_float(v);
}

// ---------- generic zero ----------
__global__ void k_zero(unsigned* p, long n) {
    long i = (long)blockIdx.x * blockDim.x + threadIdx.x;
    if (i < n) p[i] = 0u;
}

// ---------- x' = x + node_emb[node_type], to f16 ----------
__global__ void k_prep_x(const float* __restrict__ x, const int* __restrict__ ntype,
                         const float* __restrict__ nemb, _Float16* __restrict__ x16, long total) {
    long i = (long)blockIdx.x * blockDim.x + threadIdx.x;
    if (i >= total) return;
    long n = i >> 7;          // /128
    int  d = (int)(i & 127);
    x16[i] = (_Float16)(x[i] + nemb[(long)ntype[n] * D_IN + d]);
}

// ---------- pack weight [K, ncol] f32 -> per-lane WMMA B fragments f16 ----------
// layout: for col-tile ct, k-step ks: 32 lanes x 16 halves contiguous.
// lane L holds column n = ct*16 + (L&15); element j holds
// k = ks*32 + (L>=16 ? 8:0) + (j&7) + (j>>3)*16   (mirrors 16-bit A layout)
__global__ void k_pack_w(const float* __restrict__ W, _Float16* __restrict__ Wp,
                         int K, int ncol) {
    long total = (long)K * ncol;
    long idx = (long)blockIdx.x * blockDim.x + threadIdx.x;
    if (idx >= total) return;
    int j    = (int)(idx & 15);
    int lane = (int)((idx >> 4) & 31);
    long rest = idx >> 9;
    int ksteps = K >> 5;
    int ks = (int)(rest % ksteps);
    int ct = (int)(rest / ksteps);
    int k = ks * 32 + ((lane >> 4) << 3) + (j & 7) + ((j >> 3) << 4);
    int n = (ct << 4) + (lane & 15);
    Wp[idx] = (_Float16)W[(long)k * ncol + n];
}

// ---------- per-edge-type attention term table: aeT[ET, HEADS] ----------
__global__ void k_ae_table(const float* __restrict__ eemb, const float* __restrict__ We,
                           const float* __restrict__ att_e, float* __restrict__ aeT) {
    __shared__ float wred[D_IN * HEADS];   // (We . att_e)[d][h]
    int t = threadIdx.x;                   // 256 threads
    for (int idx = t; idx < D_IN * HEADS; idx += 256) {
        int d = idx >> 2, h = idx & 3;
        float s = 0.f;
        for (int c = 0; c < HID; ++c)
            s += We[(long)d * HC + h * HID + c] * att_e[h * HID + c];
        wred[idx] = s;
    }
    __syncthreads();
    if (t < ET_N * HEADS) {
        int et = t >> 2, h = t & 3;
        float s = 0.f;
        for (int d = 0; d < D_IN; ++d)
            s += eemb[(long)et * D_IN + d] * wred[d * 4 + h];
        aeT[t] = s;
    }
}

// ---------- WMMA GEMM: C[nrows,ncol] = A[nrows,K](f16) * Bp(packed f16) (+bias) ----------
// One wave computes a 16 x (16*CTW) output strip. Per k-step: one shared A
// fragment + CTW B fragments held in DISTINCT registers (loaded up-front so
// the b128 loads clause together and overlap the CTW WMMA issues), then CTW
// v_wmma_f32_16x16x32_f16.
template <int CTW>
__global__ void k_gemm_wmma(const _Float16* __restrict__ A, const _Float16* __restrict__ Bp,
                            float* __restrict__ C, const float* __restrict__ bias,
                            int nrows, int K, int ncol) {
    int lane = threadIdx.x & 31;
    int gw   = blockIdx.x * (blockDim.x >> 5) + (threadIdx.x >> 5);
    int colTiles  = ncol >> 4;
    int colGroups = colTiles / CTW;
    int strips = (nrows >> 4) * colGroups;
    if (gw >= strips) return;                      // wave-uniform: EXEC stays all-1
    int rt  = gw / colGroups;
    int cg  = gw - rt * colGroups;
    int ct0 = cg * CTW;
    int ksteps = K >> 5;
    int m  = (rt << 4) + (lane & 15);
    int kb = (lane >> 4) << 3;

    v8f acc[CTW];
#pragma unroll
    for (int i = 0; i < CTW; ++i) acc[i] = (v8f){0.f,0.f,0.f,0.f,0.f,0.f,0.f,0.f};

    const _Float16* arow = A + (long)m * K + kb;
    const _Float16* bbase[CTW];
#pragma unroll
    for (int i = 0; i < CTW; ++i)
        bbase[i] = Bp + ((long)((ct0 + i) * ksteps) * 32 + lane) * 16;

    for (int ks = 0; ks < ksteps; ++ks) {
        union { v16h v; half8 h2[2]; } a;
        a.h2[0] = *(const half8*)(arow);
        a.h2[1] = *(const half8*)(arow + 16);
        v16h b[CTW];
#pragma unroll
        for (int i = 0; i < CTW; ++i) {        // all B loads issued before any WMMA
            b[i] = *(const v16h*)(bbase[i]);
            bbase[i] += 32 * 16;
        }
#pragma unroll
        for (int i = 0; i < CTW; ++i)
            acc[i] = __builtin_amdgcn_wmma_f32_16x16x32_f16(false, a.v, false, b[i],
                                                            (short)0, acc[i], false, false);
        arow += 32;
    }

    int rbase = (rt << 4) + ((lane >> 4) << 3);
#pragma unroll
    for (int i = 0; i < CTW; ++i) {
        int col = ((ct0 + i) << 4) + (lane & 15);
        float badd = bias ? bias[col] : 0.f;
#pragma unroll
        for (int r = 0; r < 8; ++r)
            C[(long)(rbase + r) * ncol + col] = acc[i][r] + badd;
    }
}

// ---------- per-node attention dots: a_src/a_dst [N, HEADS] ----------
__global__ void k_attn_dots(const float* __restrict__ xl, const float* __restrict__ att_src,
                            const float* __restrict__ att_dst, float* __restrict__ asrc,
                            float* __restrict__ adst, long NH) {
    long idx = (long)blockIdx.x * blockDim.x + threadIdx.x;
    if (idx >= NH) return;
    long n = idx >> 2;
    int  h = (int)(idx & 3);
    const float* row = xl + n * HC + h * HID;
    float s1 = 0.f, s2 = 0.f;
    for (int c = 0; c < HID; ++c) {
        float v = row[c];
        s1 += v * att_src[h * HID + c];
        s2 += v * att_dst[h * HID + c];
    }
    asrc[idx] = s1;
    adst[idx] = s2;
}

// ---------- edge pass 1: alpha = lrelu(a_src[s]+a_dst[d]+a_e); segment-max ----------
__global__ void k_edge_alpha(const int* __restrict__ eidx, const int* __restrict__ etype,
                             const float* __restrict__ asrc, const float* __restrict__ adst,
                             const float* __restrict__ aeT, float* __restrict__ alphaB,
                             unsigned* __restrict__ menc, int E, int ETOT) {
    int e = blockIdx.x * blockDim.x + threadIdx.x;
    if (e >= ETOT) return;
    int s, d;
    float ae[HEADS] = {0.f, 0.f, 0.f, 0.f};
    if (e < E) {
        s = eidx[e];
        d = eidx[E + e];
        int t = etype[e];
#pragma unroll
        for (int h = 0; h < HEADS; ++h) ae[h] = aeT[t * HEADS + h];
    } else {
        s = d = e - E;   // self loop, zero edge feature
    }
#pragma unroll
    for (int h = 0; h < HEADS; ++h) {
        float a = asrc[(long)s * HEADS + h] + adst[(long)d * HEADS + h] + ae[h];
        a = (a > 0.f) ? a : SLOPE * a;
        alphaB[(long)e * HEADS + h] = a;
        atomicMax(&menc[(long)d * HEADS + h], f2ord(a));
    }
}

// ---------- edge pass 2: ea = exp(alpha - max); segment-sum ----------
__global__ void k_edge_expsum(const int* __restrict__ eidx, float* __restrict__ alphaB,
                              const unsigned* __restrict__ menc, float* __restrict__ denom,
                              int E, int ETOT) {
    int e = blockIdx.x * blockDim.x + threadIdx.x;
    if (e >= ETOT) return;
    int d = (e < E) ? eidx[E + e] : (e - E);
#pragma unroll
    for (int h = 0; h < HEADS; ++h) {
        float m  = ord2f(menc[(long)d * HEADS + h]);
        float ea = __expf(alphaB[(long)e * HEADS + h] - m);
        alphaB[(long)e * HEADS + h] = ea;
        atomicAdd(&denom[(long)d * HEADS + h], ea);
    }
}

// ---------- edge pass 3: agg[dst] += xl[src] * a ; one wave per edge ----------
__global__ void k_edge_aggregate(const int* __restrict__ eidx, const float* __restrict__ xl,
                                 const float* __restrict__ ea, const float* __restrict__ denom,
                                 float* __restrict__ agg, int E, int ETOT) {
    int gw = blockIdx.x * (blockDim.x >> 5) + (threadIdx.x >> 5);
    if (gw >= ETOT) return;
    int lane = threadIdx.x & 31;
    int s, d;
    if (gw < E) { s = eidx[gw]; d = eidx[E + gw]; } else { s = d = gw - E; }
    int c0 = lane << 3;          // 8 channels per lane, all in one head (8 | 64)
    int h  = c0 >> 6;
    float a = ea[(long)gw * HEADS + h] / (denom[(long)d * HEADS + h] + 1e-16f);
    const float4* xs = (const float4*)(xl + (long)s * HC + c0);
    float4 v0 = xs[0], v1 = xs[1];
    float* od = agg + (long)d * HC + c0;
    atomicAdd(od + 0, v0.x * a); atomicAdd(od + 1, v0.y * a);
    atomicAdd(od + 2, v0.z * a); atomicAdd(od + 3, v0.w * a);
    atomicAdd(od + 4, v1.x * a); atomicAdd(od + 5, v1.y * a);
    atomicAdd(od + 6, v1.z * a); atomicAdd(od + 7, v1.w * a);
}

// ---------- batchnorm column stats (blockDim == C) ----------
__global__ void k_bn_stats(const float* __restrict__ in, float* __restrict__ sum,
                           float* __restrict__ sumsq, int N, int rowsPerBlock) {
    int C = blockDim.x, col = threadIdx.x;
    int r0 = blockIdx.x * rowsPerBlock;
    int r1 = r0 + rowsPerBlock; if (r1 > N) r1 = N;
    float s = 0.f, q = 0.f;
    for (int r = r0; r < r1; ++r) {
        float v = in[(long)r * C + col];
        s += v; q += v * v;
    }
    atomicAdd(&sum[col], s);
    atomicAdd(&sumsq[col], q);
}

// ---------- batchnorm apply + ReLU -> f16 ----------
__global__ void k_bn_apply_f16(const float* __restrict__ in, const float* __restrict__ sum,
                               const float* __restrict__ sumsq, const float* __restrict__ g,
                               const float* __restrict__ be, _Float16* __restrict__ out16,
                               long total, int C, float invN) {
    long idx = (long)blockIdx.x * blockDim.x + threadIdx.x;
    if (idx >= total) return;
    int c = (int)(idx % C);
    float mu  = sum[c] * invN;
    float var = sumsq[c] * invN - mu * mu;
    float y = (in[idx] - mu) * __frsqrt_rn(var + BN_EPS) * g[c] + be[c];
    out16[idx] = (_Float16)(y > 0.f ? y : 0.f);
}

// ---------- mean over heads: [N,4,64] -> [N,64] ----------
__global__ void k_head_mean(const float* __restrict__ agg, float* __restrict__ h2, long total) {
    long idx = (long)blockIdx.x * blockDim.x + threadIdx.x;
    if (idx >= total) return;
    long n = idx >> 6;
    int  c = (int)(idx & 63);
    const float* r = agg + n * HC + c;
    h2[idx] = 0.25f * (r[0] + r[64] + r[128] + r[192]);
}

// =====================================================================

extern "C" void kernel_launch(void* const* d_in, const int* in_sizes, int n_in,
                              void* d_out, int out_size, void* d_ws, size_t ws_size,
                              hipStream_t stream) {
    const float* x        = (const float*)d_in[0];
    const int*   eidx     = (const int*)  d_in[1];
    const int*   etype    = (const int*)  d_in[2];
    const int*   ntype    = (const int*)  d_in[3];
    const float* node_emb = (const float*)d_in[4];
    const float* edge_emb = (const float*)d_in[5];
    const float* W1       = (const float*)d_in[6];
    const float* We1      = (const float*)d_in[7];
    const float* att_src1 = (const float*)d_in[8];
    const float* att_dst1 = (const float*)d_in[9];
    const float* att_e1   = (const float*)d_in[10];
    /* b1 (d_in[11]) cancels under BN mean-subtraction */
    const float* g1       = (const float*)d_in[12];
    const float* be1      = (const float*)d_in[13];
    const float* W2       = (const float*)d_in[14];
    const float* We2      = (const float*)d_in[15];
    const float* att_src2 = (const float*)d_in[16];
    const float* att_dst2 = (const float*)d_in[17];
    const float* att_e2   = (const float*)d_in[18];
    /* b2 (d_in[19]) cancels under BN mean-subtraction */
    const float* g2       = (const float*)d_in[20];
    const float* be2      = (const float*)d_in[21];
    const float* Wf       = (const float*)d_in[22];
    const float* bf       = (const float*)d_in[23];

    const int N    = in_sizes[0] / D_IN;
    const int E    = in_sizes[1] / 2;
    const int ETOT = E + N;

    // ---- workspace carve-out ----
    char* p = (char*)d_ws;
    auto alloc = [&](size_t bytes) -> char* {
        char* r = p; p += (bytes + 255) & ~(size_t)255; return r;
    };
    _Float16* x16   = (_Float16*)alloc((size_t)N * D_IN * 2);
    _Float16* w1p   = (_Float16*)alloc((size_t)D_IN * HC * 2);
    _Float16* w2p   = (_Float16*)alloc((size_t)HC * HC * 2);
    _Float16* wfp   = (_Float16*)alloc((size_t)HID * OUT_F * 2);
    float*    aeT1  = (float*)   alloc((size_t)ET_N * HEADS * 4);
    float*    aeT2  = (float*)   alloc((size_t)ET_N * HEADS * 4);
    float*    xl    = (float*)   alloc((size_t)N * HC * 4);     // reused both layers
    float*    agg   = (float*)   alloc((size_t)N * HC * 4);     // reused both layers
    float*    asrc  = (float*)   alloc((size_t)N * HEADS * 4);
    float*    adst  = (float*)   alloc((size_t)N * HEADS * 4);
    unsigned* menc  = (unsigned*)alloc((size_t)N * HEADS * 4);
    float*    denom = (float*)   alloc((size_t)N * HEADS * 4);
    float*    eab   = (float*)   alloc((size_t)ETOT * HEADS * 4); // alpha then ea
    float*    bnS   = (float*)   alloc((size_t)512 * 4);          // sum(256)+sumsq(256)
    _Float16* h1n   = (_Float16*)alloc((size_t)N * HC * 2);
    float*    h2    = (float*)   alloc((size_t)N * HID * 4);
    _Float16* h2n   = (_Float16*)alloc((size_t)N * HID * 2);

    auto cdiv = [](long a, long b) -> int { return (int)((a + b - 1) / b); };
    const int T = 256;
    const float invN = 1.0f / (float)N;

    // ---- prep: features, packed weights, edge-type tables ----
    k_prep_x<<<cdiv((long)N * D_IN, T), T, 0, stream>>>(x, ntype, node_emb, x16, (long)N * D_IN);
    k_pack_w<<<cdiv((long)D_IN * HC, T), T, 0, stream>>>(W1, w1p, D_IN, HC);
    k_pack_w<<<cdiv((long)HC * HC, T), T, 0, stream>>>(W2, w2p, HC, HC);
    k_pack_w<<<cdiv((long)HID * OUT_F, T), T, 0, stream>>>(Wf, wfp, HID, OUT_F);
    k_ae_table<<<1, 256, 0, stream>>>(edge_emb, We1, att_e1, aeT1);
    k_ae_table<<<1, 256, 0, stream>>>(edge_emb, We2, att_e2, aeT2);

    // ================= layer 1 =================
    {
        int strips = (N >> 4) * ((HC >> 4) / 4);   // 16x64 strip per wave
        k_gemm_wmma<4><<<cdiv(strips, 8), T, 0, stream>>>(x16, w1p, xl, nullptr, N, D_IN, HC);
    }
    k_attn_dots<<<cdiv((long)N * HEADS, T), T, 0, stream>>>(xl, att_src1, att_dst1, asrc, adst, (long)N * HEADS);

    k_zero<<<cdiv((long)N * HEADS, T), T, 0, stream>>>(menc, (long)N * HEADS);
    k_zero<<<cdiv((long)N * HEADS, T), T, 0, stream>>>((unsigned*)denom, (long)N * HEADS);
    k_zero<<<cdiv((long)N * HC, T), T, 0, stream>>>((unsigned*)agg, (long)N * HC);

    k_edge_alpha<<<cdiv(ETOT, T), T, 0, stream>>>(eidx, etype, asrc, adst, aeT1, eab, menc, E, ETOT);
    k_edge_expsum<<<cdiv(ETOT, T), T, 0, stream>>>(eidx, eab, menc, denom, E, ETOT);
    k_edge_aggregate<<<cdiv(ETOT, 8), T, 0, stream>>>(eidx, xl, eab, denom, agg, E, ETOT);

    k_zero<<<cdiv(512L, T), T, 0, stream>>>((unsigned*)bnS, 512L);
    k_bn_stats<<<cdiv(N, 128), HC, 0, stream>>>(agg, bnS, bnS + 256, N, 128);
    k_bn_apply_f16<<<cdiv((long)N * HC, T), T, 0, stream>>>(agg, bnS, bnS + 256, g1, be1,
                                                            h1n, (long)N * HC, HC, invN);

    // ================= layer 2 =================
    {
        int strips = (N >> 4) * ((HC >> 4) / 4);
        k_gemm_wmma<4><<<cdiv(strips, 8), T, 0, stream>>>(h1n, w2p, xl, nullptr, N, HC, HC);
    }
    k_attn_dots<<<cdiv((long)N * HEADS, T), T, 0, stream>>>(xl, att_src2, att_dst2, asrc, adst, (long)N * HEADS);

    k_zero<<<cdiv((long)N * HEADS, T), T, 0, stream>>>(menc, (long)N * HEADS);
    k_zero<<<cdiv((long)N * HEADS, T), T, 0, stream>>>((unsigned*)denom, (long)N * HEADS);
    k_zero<<<cdiv((long)N * HC, T), T, 0, stream>>>((unsigned*)agg, (long)N * HC);

    k_edge_alpha<<<cdiv(ETOT, T), T, 0, stream>>>(eidx, etype, asrc, adst, aeT2, eab, menc, E, ETOT);
    k_edge_expsum<<<cdiv(ETOT, T), T, 0, stream>>>(eidx, eab, menc, denom, E, ETOT);
    k_edge_aggregate<<<cdiv(ETOT, 8), T, 0, stream>>>(eidx, xl, eab, denom, agg, E, ETOT);

    k_head_mean<<<cdiv((long)N * HID, T), T, 0, stream>>>(agg, h2, (long)N * HID);
    k_zero<<<cdiv(128L, T), T, 0, stream>>>((unsigned*)bnS, 128L);
    k_bn_stats<<<cdiv(N, 128), HID, 0, stream>>>(h2, bnS, bnS + 64, N, 128);
    k_bn_apply_f16<<<cdiv((long)N * HID, T), T, 0, stream>>>(h2, bnS, bnS + 64, g2, be2,
                                                             h2n, (long)N * HID, HID, invN);

    // ================= final projection (WMMA, fused bias) =================
    {
        int strips = (N >> 4) * ((OUT_F >> 4) / 2);  // 16x32 strip per wave
        k_gemm_wmma<2><<<cdiv(strips, 8), T, 0, stream>>>(h2n, wfp, (float*)d_out, bf, N, HID, OUT_F);
    }
}
